// EdgeGNN_3573412790510
// MI455X (gfx1250) — matrix operand order repaired and checked by reading, not compiled, not run on hardware
//
#include <hip/hip_runtime.h>
#include <hip/hip_bf16.h>

#define NNODES 10000
#define NEDGES 320000
#define IN_DIM 256
#define HID 256
#define EDGE_DIM 64

typedef __bf16 bf16_t;
typedef bf16_t v16bf __attribute__((ext_vector_type(16)));
typedef float  v8f  __attribute__((ext_vector_type(8)));

// ---- helpers ----------------------------------------------------------------

__device__ __forceinline__ bf16_t f2bf(float f) {
    unsigned u = __float_as_uint(f);
    unsigned r = (u + 0x7FFFu + ((u >> 16) & 1u)) >> 16;   // round-to-nearest-even
    unsigned short s = (unsigned short)r;
    return __builtin_bit_cast(bf16_t, s);
}

// ---- small utility kernels --------------------------------------------------

__global__ void k_zero_u32(unsigned* __restrict__ p, int n) {
    int i = blockIdx.x * blockDim.x + threadIdx.x;
    if (i < n) p[i] = 0u;
}

// strided f32 -> bf16 cast: dst[r*dstride + dcoloff + c] = bf16(src[r*sstride + c])
__global__ void k_cast_bf(const float* __restrict__ src, bf16_t* __restrict__ dst,
                          int rows, int cols, int sstride, int dstride, int dcoloff) {
    int i = blockIdx.x * blockDim.x + threadIdx.x;
    if (i >= rows * cols) return;
    int r = i / cols, c = i - r * cols;
    dst[(size_t)r * dstride + dcoloff + c] = f2bf(src[(size_t)r * sstride + c]);
}

__global__ void k_degree(const int* __restrict__ dst, int* __restrict__ cnt, int nE) {
    int e = blockIdx.x * blockDim.x + threadIdx.x;
    if (e < nE) atomicAdd(&cnt[dst[e]], 1);
}

// scatter-add rows of xin[src[e]] into agg[dst[e]] (256 floats/row, 4 per thread)
__global__ void k_scatter(const float* __restrict__ xin, const int* __restrict__ src,
                          const int* __restrict__ dst, float* __restrict__ agg, int nE) {
    int idx = blockIdx.x * blockDim.x + threadIdx.x;
    if (idx >= nE * 64) return;
    int e  = idx >> 6;
    int f4 = (idx & 63) << 2;
    const float4 v = *(const float4*)(xin + (size_t)src[e] * 256 + f4);
    float* o = agg + (size_t)dst[e] * 256 + f4;
    atomicAdd(o + 0, v.x); atomicAdd(o + 1, v.y);
    atomicAdd(o + 2, v.z); atomicAdd(o + 3, v.w);
}

// mean = agg / max(cnt,1), stored as bf16 into the A matrix (left half)
__global__ void k_mean(const float* __restrict__ agg, const int* __restrict__ cnt,
                       bf16_t* __restrict__ dstA, int dstride, int dcoloff, int nN) {
    int idx = blockIdx.x * blockDim.x + threadIdx.x;
    if (idx >= nN * 64) return;
    int n  = idx >> 6;
    int f4 = (idx & 63) << 2;
    float inv = 1.0f / fmaxf((float)cnt[n], 1.0f);
    float4 v = *(const float4*)(agg + (size_t)n * 256 + f4);
    bf16_t* o = dstA + (size_t)n * dstride + dcoloff + f4;
    o[0] = f2bf(v.x * inv); o[1] = f2bf(v.y * inv);
    o[2] = f2bf(v.z * inv); o[3] = f2bf(v.w * inv);
}

// ---- node GEMM: out[M,256] = A[M,K](bf16) @ W[256,K](bf16)^T + bias ---------
// block = 128 threads (4 waves); each wave: 16 rows x 64 cols (4 wmma N-tiles)
__global__ void __launch_bounds__(128)
k_gemm_node(const bf16_t* __restrict__ A, const bf16_t* __restrict__ W,
            const float* __restrict__ bias, float* __restrict__ outf,
            bf16_t* __restrict__ outb, int obStride, int obColOff,
            int K, int relu) {
    const int lane = threadIdx.x & 31;
    const int wave = threadIdx.x >> 5;
    const int rowBase = blockIdx.x * 16;
    const int colWave = wave * 64;
    const int ln = lane & 15;
    const int hi = lane >> 4;                 // 0: lanes 0-15, 1: lanes 16-31

    v8f c[4] = {};
    const bf16_t* aptr = A + (size_t)(rowBase + ln) * K;

    for (int kb = 0; kb < K; kb += 32) {
        // A 16x32 bf16: lane row = ln; K chunks {k0..k0+7, k0+16..k0+23}, k0 = kb + 8*hi
        v16bf a;
        int k0 = kb + hi * 8;
        *(uint4*)&a       = *(const uint4*)(aptr + k0);
        *((uint4*)&a + 1) = *(const uint4*)(aptr + k0 + 16);
#pragma unroll
        for (int nt = 0; nt < 4; ++nt) {
            // B 32x16 bf16: lane col n = colWave+16*nt+ln; K chunks {k0..k0+7, k0+16..k0+23}, k0 = kb + 8*hi
            int n = colWave + nt * 16 + ln;
            const bf16_t* bptr = W + (size_t)n * K + kb + hi * 8;
            v16bf b;
            *(uint4*)&b       = *(const uint4*)(bptr);
            *((uint4*)&b + 1) = *(const uint4*)(bptr + 16);
            c[nt] = __builtin_amdgcn_wmma_f32_16x16x32_bf16(
                false, a, false, b, (short)0, c[nt], false, false);
        }
    }
    // epilogue: bias (+relu), dual store f32 + bf16
#pragma unroll
    for (int nt = 0; nt < 4; ++nt) {
        int n = colWave + nt * 16 + ln;
        float bv = bias[n];
#pragma unroll
        for (int v = 0; v < 8; ++v) {
            int m = rowBase + v + hi * 8;     // D layout: lanes<16 -> m=v, lanes>=16 -> m=v+8
            float val = c[nt][v] + bv;
            if (relu) val = fmaxf(val, 0.0f);
            if (outf) outf[(size_t)m * 256 + n] = val;
            outb[(size_t)m * obStride + obColOff + n] = f2bf(val);
        }
    }
}

// ---- fused edge MLP: logits[e] = w2 . relu([h[src]|h[dst]|ea] @ W1^T + b1) + b2
// block = 256 (8 waves); wave -> 16 edges, all 256 hidden cols (16 accum tiles)
// mlp_w1 (576x256 bf16, 288KB) is DMA'd once per block into the 320KB WGP LDS
// with gfx1250 async-to-LDS loads; rows padded to 584 elems (1168B) so the 16
// B-lanes hit distinct banks (36-word stride is a full permutation mod 64).
#define SW_STRIDE 584
__global__ void __launch_bounds__(256)
k_gemm_edge(const bf16_t* __restrict__ h2b, const float* __restrict__ ea,
            const bf16_t* __restrict__ W, const float* __restrict__ b1,
            const float* __restrict__ w2, const float* __restrict__ b2,
            const int* __restrict__ src, const int* __restrict__ dst,
            float* __restrict__ out) {
    __shared__ __align__(16) bf16_t sW[256 * SW_STRIDE];   // 299,008 bytes

    const int lane = threadIdx.x & 31;
    const int wave = threadIdx.x >> 5;
    const int tile = blockIdx.x * 8 + wave;
    const int eBase = tile * 16;
    const int ln = lane & 15;
    const int hi = lane >> 4;

    // ---- stage W into LDS via async DMA (72 x 16B chunks per thread) ----
    {
        // sW is the only static LDS allocation -> group-segment offset 0.
        for (int i = 0; i < 72; ++i) {
            int c = i * 256 + (int)threadIdx.x;        // 0 .. 18431
            int row = c / 72;
            int k8  = c - row * 72;                    // 16B chunk within row
            unsigned ldsOff = (unsigned)(row * (SW_STRIDE * 2) + k8 * 16);
            unsigned long long g =
                (unsigned long long)(uintptr_t)W + (unsigned long long)(row * 1152 + k8 * 16);
            asm volatile("global_load_async_to_lds_b128 %0, %1, off"
                         :: "v"(ldsOff), "v"(g) : "memory");
        }
        asm volatile("s_wait_asynccnt 0x0" ::: "memory");
        __syncthreads();
    }

    const int e  = eBase + ln;                // A row handled by this lane
    const int es = src[e];
    const int ed = dst[e];

    float b1r[16], w2r[16];                   // per-lane column constants: n = 16*t + ln
#pragma unroll
    for (int t = 0; t < 16; ++t) { b1r[t] = b1[t * 16 + ln]; w2r[t] = w2[t * 16 + ln]; }

    v8f c[16] = {};
    for (int kb = 0; kb < 576; kb += 32) {
        int k0 = kb + hi * 8;
        v16bf a;
        if (k0 < 256) {                       // h[src] region
            const bf16_t* p = h2b + (size_t)es * 256 + k0;
            *(uint4*)&a       = *(const uint4*)p;
            *((uint4*)&a + 1) = *(const uint4*)(p + 16);
        } else if (k0 < 512) {                // h[dst] region
            const bf16_t* p = h2b + (size_t)ed * 256 + (k0 - 256);
            *(uint4*)&a       = *(const uint4*)p;
            *((uint4*)&a + 1) = *(const uint4*)(p + 16);
        } else {                              // edge_attr region (f32 -> bf16 on the fly)
            const float* p = ea + (size_t)e * 64 + (k0 - 512);
#pragma unroll
            for (int j = 0; j < 8; ++j) { a[j] = f2bf(p[j]); a[8 + j] = f2bf(p[16 + j]); }
        }
#pragma unroll
        for (int t = 0; t < 16; ++t) {
            const bf16_t* bptr = sW + (t * 16 + ln) * SW_STRIDE + kb + hi * 8;
            v16bf b;
            *(uint4*)&b       = *(const uint4*)bptr;
            *((uint4*)&b + 1) = *(const uint4*)(bptr + 16);
            c[t] = __builtin_amdgcn_wmma_f32_16x16x32_bf16(
                false, a, false, b, (short)0, c[t], false, false);
        }
    }

    // fused second layer: relu(h + b1) . w2, reduced over n
    float acc[8];
#pragma unroll
    for (int v = 0; v < 8; ++v) acc[v] = 0.0f;
#pragma unroll
    for (int t = 0; t < 16; ++t)
#pragma unroll
        for (int v = 0; v < 8; ++v) {
            float hv = fmaxf(c[t][v] + b1r[t], 0.0f);
            acc[v] += hv * w2r[t];
        }
    // reduce across the 16 lanes of each half (halves hold disjoint m ranges)
#pragma unroll
    for (int v = 0; v < 8; ++v) {
        for (int m = 1; m < 16; m <<= 1) acc[v] += __shfl_xor(acc[v], m, 32);
    }
    float bb = b2[0];
    if (ln == 0) {
#pragma unroll
        for (int v = 0; v < 8; ++v) out[eBase + hi * 8 + v] = acc[v] + bb;
    }
}

// ---- launch -----------------------------------------------------------------

extern "C" void kernel_launch(void* const* d_in, const int* in_sizes, int n_in,
                              void* d_out, int out_size, void* d_ws, size_t ws_size,
                              hipStream_t stream) {
    const float* x    = (const float*)d_in[0];
    const float* ea   = (const float*)d_in[1];
    const float* c1wl = (const float*)d_in[2];
    const float* c1bl = (const float*)d_in[3];
    const float* c1wr = (const float*)d_in[4];
    const float* c2wl = (const float*)d_in[5];
    const float* c2bl = (const float*)d_in[6];
    const float* c2wr = (const float*)d_in[7];
    const float* mw1  = (const float*)d_in[8];
    const float* mb1  = (const float*)d_in[9];
    const float* mw2  = (const float*)d_in[10];
    const float* mb2  = (const float*)d_in[11];
    const int*   ei   = (const int*)d_in[12];
    const int* srcp = ei;
    const int* dstp = ei + NEDGES;
    float* out = (float*)d_out;

    char* ws = (char*)d_ws;
    size_t off = 0;
    auto alloc = [&](size_t bytes) -> void* {
        void* p = ws + off;
        off += (bytes + 255) & ~(size_t)255;
        return p;
    };
    int*    cnt = (int*)   alloc((size_t)NNODES * 4);
    float*  agg = (float*) alloc((size_t)NNODES * 256 * 4);
    bf16_t* A1  = (bf16_t*)alloc((size_t)NNODES * 512 * 2);
    bf16_t* A2  = (bf16_t*)alloc((size_t)NNODES * 512 * 2);
    float*  h1  = (float*) alloc((size_t)NNODES * 256 * 4);
    bf16_t* h2b = (bf16_t*)alloc((size_t)NNODES * 256 * 2);
    bf16_t* Wc1 = (bf16_t*)alloc((size_t)256 * 512 * 2);
    bf16_t* Wc2 = (bf16_t*)alloc((size_t)256 * 512 * 2);
    bf16_t* W1e = (bf16_t*)alloc((size_t)256 * 576 * 2);
    (void)ws_size; (void)in_sizes; (void)n_in; (void)out_size;

    const int T = 256;
    auto blocks = [](long long n, int t) { return (int)((n + t - 1) / t); };

    // zero scatter accumulators
    k_zero_u32<<<blocks(NNODES, T), T, 0, stream>>>((unsigned*)cnt, NNODES);
    k_zero_u32<<<blocks((long long)NNODES * 256, T), T, 0, stream>>>((unsigned*)agg, NNODES * 256);

    // pack bf16 operands: A1 right half = x; Wc = [w_l ; w_r]; W1e = mlp_w1
    k_cast_bf<<<blocks((long long)NNODES * 256, T), T, 0, stream>>>(x, A1, NNODES, 256, 256, 512, 256);
    k_cast_bf<<<blocks(256 * 256, T), T, 0, stream>>>(c1wl, Wc1, 256, 256, 256, 512, 0);
    k_cast_bf<<<blocks(256 * 256, T), T, 0, stream>>>(c1wr, Wc1, 256, 256, 256, 512, 256);
    k_cast_bf<<<blocks(256 * 256, T), T, 0, stream>>>(c2wl, Wc2, 256, 256, 256, 512, 0);
    k_cast_bf<<<blocks(256 * 256, T), T, 0, stream>>>(c2wr, Wc2, 256, 256, 256, 512, 256);
    k_cast_bf<<<blocks(256 * 576, T), T, 0, stream>>>(mw1, W1e, 256, 576, 576, 576, 0);

    // layer 1 aggregation (mean)
    k_degree<<<blocks(NEDGES, T), T, 0, stream>>>(dstp, cnt, NEDGES);
    k_scatter<<<blocks((long long)NEDGES * 64, T), T, 0, stream>>>(x, srcp, dstp, agg, NEDGES);
    k_mean<<<blocks((long long)NNODES * 64, T), T, 0, stream>>>(agg, cnt, A1, 512, 0, NNODES);

    // layer 1 GEMM: h1 = relu([mean|x] @ [wl;wr]^T + bl); bf16 copy into A2 right half
    k_gemm_node<<<NNODES / 16, 128, 0, stream>>>(A1, Wc1, c1bl, h1, A2, 512, 256, 512, 1);

    // layer 2 aggregation (counts are identical; only re-zero agg)
    k_zero_u32<<<blocks((long long)NNODES * 256, T), T, 0, stream>>>((unsigned*)agg, NNODES * 256);
    k_scatter<<<blocks((long long)NEDGES * 64, T), T, 0, stream>>>(h1, srcp, dstp, agg, NEDGES);
    k_mean<<<blocks((long long)NNODES * 64, T), T, 0, stream>>>(agg, cnt, A2, 512, 0, NNODES);

    // layer 2 GEMM: h2 (bf16 only, no relu)
    k_gemm_node<<<NNODES / 16, 128, 0, stream>>>(A2, Wc2, c2bl, nullptr, h2b, 256, 0, 512, 0);

    // fused edge MLP -> logits (8 tiles of 16 edges per 256-thread block)
    k_gemm_edge<<<NEDGES / 16 / 8, 256, 0, stream>>>(h2b, ea, W1e, mb1, mw2, mb2, srcp, dstp, out);
}
